// LSTM_55061480735178
// MI455X (gfx1250) — compile-verified, matrix operand order checked by reading, not run on hardware
//
#include <hip/hip_runtime.h>
#include <hip/hip_bf16.h>
#include <stddef.h>

// ---------------------------------------------------------------------------
// Sizes (match reference): B=128, T=512, D=H=256, L=2, 4H=1024
// ---------------------------------------------------------------------------
#define BB   128
#define TT   512
#define HH   256
#define G4   1024      // 4*H
#define MR   65536     // B*T

typedef __attribute__((ext_vector_type(16))) __bf16 v16bf;
typedef __attribute__((ext_vector_type(8)))  float  v8f;
typedef unsigned int u32x4 __attribute__((ext_vector_type(4)));
typedef int          i32x4 __attribute__((ext_vector_type(4)));
typedef int          i32x8 __attribute__((ext_vector_type(8)));

union V16 { v16bf v; u32x4 q[2]; };

#if __has_builtin(__builtin_amdgcn_tensor_load_to_lds) && __has_builtin(__builtin_amdgcn_s_wait_tensorcnt)
#define HAVE_TDM 1
#else
#define HAVE_TDM 0
#endif

static __device__ inline v8f wmma_bf16(v16bf a, v16bf b, v8f c) {
  return __builtin_amdgcn_wmma_f32_16x16x32_bf16(false, a, false, b, (short)0, c, false, false);
}
static __device__ inline float sigmoidf_(float x) {
  return 1.0f / (1.0f + __expf(-x));
}
// uniform base + 32-bit element offset -> saddr+voffset global loads
static __device__ inline void load16(V16& d, const __bf16* __restrict__ base, unsigned eoff) {
  d.q[0] = *(const u32x4*)(base + eoff);
  d.q[1] = *(const u32x4*)(base + eoff + 16);
}

// ---------------------------------------------------------------------------
// fp32 -> bf16 conversion (grid-stride)
// ---------------------------------------------------------------------------
__global__ void __launch_bounds__(256) cvt_f32_bf16(const float* __restrict__ s,
                                                    __bf16* __restrict__ d,
                                                    long long n) {
  long long i = (long long)blockIdx.x * blockDim.x + threadIdx.x;
  long long st = (long long)gridDim.x * blockDim.x;
  for (; i < n; i += st) d[i] = (__bf16)s[i];
}

__global__ void __launch_bounds__(256) bias_sum(const float* __restrict__ bih,
                                                const float* __restrict__ bhh,
                                                float* __restrict__ bsum, int n) {
  int i = blockIdx.x * blockDim.x + threadIdx.x;
  if (i < n) bsum[i] = bih[i] + bhh[i];
}

// ---------------------------------------------------------------------------
// Input projection GEMM: G = X (bf16, [MR,256]) * W^T (bf16, [1024,256]) + bsum
// Workgroup = 256 thr (8 waves), WG tile 128(M) x 128(N).
// Wave = 32(M) x 64(N): 2 M-subtiles x 4 N-tiles = 8 accum tiles -> 8 WMMAs
// per 12 b128 loads. Full K-chunk double buffering; launch_bounds(256,1)
// gives the RA enough VGPRs to keep both chunks live.
// ---------------------------------------------------------------------------
__global__ void __launch_bounds__(256, 1) gemm_xg(const __bf16* __restrict__ X0,
                                                  const __bf16* __restrict__ X1,
                                                  const __bf16* __restrict__ W,
                                                  const float*  __restrict__ bsum,
                                                  __bf16* __restrict__ Gout0,
                                                  __bf16* __restrict__ Gout1) {
  const __bf16* X = blockIdx.z ? X1 : X0;
  __bf16* Gout    = blockIdx.z ? Gout1 : Gout0;

  const int wave = threadIdx.x >> 5;
  const int lane = threadIdx.x & 31;
  const int half = lane >> 4;
  const int lm   = lane & 15;
  const int mt2  = wave >> 1;                 // 0..3 : which 32-row stripe
  const int ns   = wave & 1;                  // 0..1 : which 64-col slice
  const int row0 = blockIdx.y * 128 + mt2 * 32;
  const int col0 = blockIdx.x * 128 + ns * 64;

  unsigned aoff[2], boff[4];
#pragma unroll
  for (int u = 0; u < 2; ++u)
    aoff[u] = (unsigned)(row0 + u * 16 + lm) * HH + half * 8;
#pragma unroll
  for (int nt = 0; nt < 4; ++nt)
    boff[nt] = (unsigned)(col0 + nt * 16 + lm) * HH + half * 8;

  v8f acc[2][4] = {};
  V16 a[2][2];                                 // [buf][u]
  V16 b[2][4];                                 // [buf][nt]

  // prologue: chunk 0 in flight
#pragma unroll
  for (int u = 0; u < 2; ++u) load16(a[0][u], X, aoff[u]);
#pragma unroll
  for (int nt = 0; nt < 4; ++nt) load16(b[0][nt], W, boff[nt]);

#pragma unroll
  for (int kc = 0; kc < 8; ++kc) {
    const int cur = kc & 1, nxt = cur ^ 1;
    if (kc < 7) {
      const unsigned kb = (unsigned)(kc + 1) * 32;
#pragma unroll
      for (int u = 0; u < 2; ++u) load16(a[nxt][u], X, aoff[u] + kb);
#pragma unroll
      for (int nt = 0; nt < 4; ++nt) load16(b[nxt][nt], W, boff[nt] + kb);
    }
#pragma unroll
    for (int u = 0; u < 2; ++u)
#pragma unroll
      for (int nt = 0; nt < 4; ++nt)
        acc[u][nt] = wmma_bf16(a[cur][u].v, b[cur][nt].v, acc[u][nt]);
  }

#pragma unroll
  for (int u = 0; u < 2; ++u)
#pragma unroll
    for (int nt = 0; nt < 4; ++nt) {
      const int n = col0 + nt * 16 + lm;
      const float bias = bsum[n];
#pragma unroll
      for (int r = 0; r < 8; ++r) {
        const unsigned m = (unsigned)(row0 + u * 16 + r + half * 8);
        Gout[m * (unsigned)G4 + (unsigned)n] = (__bf16)(acc[u][nt][r] + bias);
      }
    }
}

// ---------------------------------------------------------------------------
// TDM: prefetch a 16 x 1024 bf16 tile (rows strided by T*4H elems) into LDS.
// ---------------------------------------------------------------------------
#if HAVE_TDM
static __device__ inline void tdm_prefetch_tile(const __bf16* gptr, unsigned lds_off) {
  unsigned long long ga = (unsigned long long)(size_t)gptr;
  u32x4 g0;
  g0[0] = 1u;                                            // count = 1 (valid)
  g0[1] = lds_off;                                       // lds_addr
  g0[2] = (unsigned)(ga & 0xFFFFFFFFu);                  // global_addr[31:0]
  g0[3] = (unsigned)((ga >> 32) & 0x01FFFFFFu) | (2u << 30); // addr[56:32] | type=2
  i32x8 g1;
  g1[0] = 0x00010000;                                    // data_size=1 (2B)
  g1[1] = (int)((unsigned)(G4 & 0xFFFF) << 16);          // tensor_dim0[15:0]
  g1[2] = (int)(16u << 16);                              // tensor_dim1[15:0]=16
  g1[3] = (int)((unsigned)G4 << 16);                     // tile_dim0 = 1024
  g1[4] = 16;                                            // tile_dim1 = 16
  g1[5] = (int)(TT * G4);                                // tensor_dim0_stride
  g1[6] = 0;
  g1[7] = 0;
  i32x4 z4 = {};
#if defined(__clang_major__) && (__clang_major__ >= 23)
  i32x8 z8 = {};
  __builtin_amdgcn_tensor_load_to_lds(g0, g1, z4, z4, z8, 0);
#else
  __builtin_amdgcn_tensor_load_to_lds(g0, g1, z4, z4, 0);
#endif
}
#endif

// ---------------------------------------------------------------------------
// LSTM recurrent scan. grid = (8 batch-tiles, 2 inputs), 256 threads (8 waves).
// Each WG owns batch rows [b0, b0+16). Wave w owns hidden cols [w*32,(w+1)*32)
// for ALL four gates -> cell update is wave-local; c lives in VGPRs.
// Gate GEMM pipelined with a 3-deep B ring (lookahead = 2 groups = 16 b128
// loads in flight behind each 4-WMMA burst). launch_bounds(256,1): the scan
// grid is tiny (16 WGs), occupancy is irrelevant, registers are not.
// Dynamic LDS: xg dbl-buf 64KB (TDM-filled), h dbl-buf 16KB.
// ---------------------------------------------------------------------------
__global__ void __launch_bounds__(256, 1) lstm_scan(const __bf16* __restrict__ XG0,
                                                    const __bf16* __restrict__ XG1,
                                                    const __bf16* __restrict__ Whh,
                                                    __bf16* __restrict__ HS0,
                                                    __bf16* __restrict__ HS1,
                                                    float* __restrict__ HN0,
                                                    float* __restrict__ HN1,
                                                    int write_seq) {
  extern __shared__ char smem[];
  __bf16* xgbuf = (__bf16*)smem;                 // 2 * 16*1024
  __bf16* hbuf  = (__bf16*)(smem + 2 * 16 * G4 * sizeof(__bf16)); // 2 * 16*256

  const int input = blockIdx.y;
  const __bf16* XG = input ? XG1 : XG0;
  __bf16* HS = input ? HS1 : HS0;
  float*  HN = input ? HN1 : HN0;
  const int b0 = blockIdx.x * 16;

  const int tid  = threadIdx.x;
  const int wave = tid >> 5;
  const int lane = tid & 31;
  const int half = lane >> 4;
  const int lm   = lane & 15;
  const int wcol = wave * 32;                    // hidden-col base of this wave

  // Loop-invariant 32-bit weight offsets: tile (kc, q, s) is at
  // Whh + woff[q][s] + kc*32 (+16 for second 16B chunk).
  unsigned woff[4][2];
#pragma unroll
  for (int q = 0; q < 4; ++q)
#pragma unroll
    for (int s = 0; s < 2; ++s)
      woff[q][s] = (unsigned)(q * HH + wcol + s * 16 + lm) * HH + half * 8;

  // zero both h buffers (h0 = 0)
  for (int i = tid; i < 2 * 16 * HH; i += 256) hbuf[i] = (__bf16)0.0f;

#if HAVE_TDM
  const unsigned xg_lds0 = (unsigned)(size_t)(void*)xgbuf;
  if (wave == 0) {
    tdm_prefetch_tile(XG + (size_t)b0 * TT * G4, xg_lds0);
    __builtin_amdgcn_s_wait_tensorcnt(0);
  }
#endif
  __syncthreads();

  v8f cst[2] = {};                               // cell state, 16 cells/lane

  for (int t = 0; t < TT; ++t) {
    const int cur = t & 1, nxt = cur ^ 1;
#if HAVE_TDM
    if (wave == 0 && (t + 1) < TT)
      tdm_prefetch_tile(XG + ((size_t)b0 * TT + (t + 1)) * G4,
                        xg_lds0 + (unsigned)(nxt * 16 * G4 * sizeof(__bf16)));
#endif
    // -------- gate GEMM: acc[q][s] += h[16x256] * Whh[gate cols]^T ---------
    const __bf16* hb = hbuf + cur * 16 * HH + lm * HH + half * 8;
    v8f acc[4][2] = {};
    V16 a[2];                                    // per-kc A, dbl-buffered
    V16 bb[3][4];                                // B ring, lookahead 2 groups

    // prologue: A chunk 0 + B groups 0 (kc0,s0) and 1 (kc0,s1)
    load16(a[0], hb, 0);
#pragma unroll
    for (int q = 0; q < 4; ++q) load16(bb[0][q], Whh, woff[q][0]);
#pragma unroll
    for (int q = 0; q < 4; ++q) load16(bb[1][q], Whh, woff[q][1]);

#pragma unroll
    for (int g = 0; g < 16; ++g) {               // group g: kc=g>>1, s=g&1
      const int kc = g >> 1, s = g & 1;
      if (g < 14) {
        const int ng = g + 2, nkc = ng >> 1, nsl = ng & 1;
        if (nsl == 0) load16(a[nkc & 1], hb, (unsigned)nkc * 32);
#pragma unroll
        for (int q = 0; q < 4; ++q)
          load16(bb[ng % 3][q], Whh, woff[q][nsl] + (unsigned)nkc * 32);
      }
#pragma unroll
      for (int q = 0; q < 4; ++q)
        acc[q][s] = wmma_bf16(a[kc & 1].v, bb[g % 3][q].v, acc[q][s]);
    }

    // -------- cell update (wave-local: this wave holds i,f,g,o) ------------
#pragma unroll
    for (int s = 0; s < 2; ++s) {
      const int n = wcol + s * 16 + lm;          // hidden unit
#pragma unroll
      for (int r = 0; r < 8; ++r) {
        const int m = r + half * 8;              // batch row within tile
        float xq[4];
#pragma unroll
        for (int q = 0; q < 4; ++q) {
#if HAVE_TDM
          xq[q] = (float)xgbuf[cur * 16 * G4 + m * G4 + q * HH + n];
#else
          xq[q] = (float)XG[((size_t)(b0 + m) * TT + t) * G4 + q * HH + n];
#endif
        }
        const float gi = sigmoidf_(acc[0][s][r] + xq[0]);
        const float gf = sigmoidf_(acc[1][s][r] + xq[1]);
        const float gg = tanhf(acc[2][s][r] + xq[2]);
        const float go = sigmoidf_(acc[3][s][r] + xq[3]);
        const float c  = gf * cst[s][r] + gi * gg;
        cst[s][r] = c;
        const float h = go * tanhf(c);
        hbuf[nxt * 16 * HH + m * HH + n] = (__bf16)h;
        if (write_seq) HS[(unsigned)((b0 + m) * TT + t) * G4 / 4 + (unsigned)n] = (__bf16)h;
        if (t == TT - 1) HN[(unsigned)(b0 + m) * HH + (unsigned)n] = h;
      }
    }
#if HAVE_TDM
    if (wave == 0 && (t + 1) < TT) __builtin_amdgcn_s_wait_tensorcnt(0);
#endif
    __syncthreads();
  }
}

// ---------------------------------------------------------------------------
// FC head: out[i, l*B+b, j] = relu(hn) . fc_w[j] + fc_b[j]   (tiny)
// ---------------------------------------------------------------------------
__global__ void __launch_bounds__(256) fc_head(const float* __restrict__ HN0,
                                               const float* __restrict__ HN1,
                                               const float* __restrict__ fcw,
                                               const float* __restrict__ fcb,
                                               float* __restrict__ out) {
  const int tid = blockIdx.x * blockDim.x + threadIdx.x;
  if (tid >= 2 * 2 * BB * 5) return;
  const int i   = tid / (2 * BB * 5);
  const int rem = tid % (2 * BB * 5);
  const int row = rem / 5;
  const int j   = rem % 5;
  const float* hn = (i ? HN1 : HN0) + (size_t)row * HH;
  float s = fcb[j];
  for (int h = 0; h < HH; ++h) {
    float v = hn[h];
    v = v > 0.0f ? v : 0.0f;
    s += v * fcw[j * HH + h];
  }
  out[(size_t)i * (2 * BB * 5) + row * 5 + j] = s;
}

// ---------------------------------------------------------------------------
// Host launcher
// ---------------------------------------------------------------------------
extern "C" void kernel_launch(void* const* d_in, const int* in_sizes, int n_in,
                              void* d_out, int out_size, void* d_ws, size_t ws_size,
                              hipStream_t stream) {
  (void)in_sizes; (void)n_in; (void)out_size; (void)ws_size;
  const float* in1 = (const float*)d_in[0];
  const float* in2 = (const float*)d_in[1];
  const float* Wih = (const float*)d_in[2];   // [2,1024,256]
  const float* Whh = (const float*)d_in[3];   // [2,1024,256]
  const float* bih = (const float*)d_in[4];   // [2,1024]
  const float* bhh = (const float*)d_in[5];
  const float* fcw = (const float*)d_in[6];   // [5,256]
  const float* fcb = (const float*)d_in[7];

  char* ws = (char*)d_ws;
  const size_t WSZ = (size_t)2 * G4 * HH;     // elems per weight set
  __bf16* wih_bf = (__bf16*)(ws + 0);                        // 1 MB
  __bf16* whh_bf = (__bf16*)(ws + ((size_t)1 << 20));        // 1 MB
  float*  bsum   = (float*)(ws + ((size_t)2 << 20));         // 8 KB
  float*  hn0    = (float*)(ws + ((size_t)2 << 20) + (64 << 10));   // 2*128*256 f32
  float*  hn1    = (float*)(ws + ((size_t)2 << 20) + (64 << 10) + (256 << 10));
  size_t base = (size_t)4 << 20;
  const size_t XSZ = (size_t)MR * HH;         // 16.7M elems (32 MB bf16)
  __bf16* xbf0 = (__bf16*)(ws + base);
  __bf16* xbf1 = xbf0 + XSZ;
  __bf16* hs0  = xbf1 + XSZ;
  __bf16* hs1  = hs0 + XSZ;
  __bf16* xg0  = hs1 + XSZ;                   // [MR,1024] bf16 = 128 MB
  __bf16* xg1  = xg0 + (size_t)MR * G4;

  // weight / input conversion
  cvt_f32_bf16<<<2048, 256, 0, stream>>>(Wih, wih_bf, (long long)WSZ);
  cvt_f32_bf16<<<2048, 256, 0, stream>>>(Whh, whh_bf, (long long)WSZ);
  cvt_f32_bf16<<<4096, 256, 0, stream>>>(in1, xbf0, (long long)XSZ);
  cvt_f32_bf16<<<4096, 256, 0, stream>>>(in2, xbf1, (long long)XSZ);
  bias_sum<<<8, 256, 0, stream>>>(bih, bhh, bsum, 2 * G4);

  const unsigned smem = 2 * 16 * G4 * sizeof(__bf16) + 2 * 16 * HH * sizeof(__bf16); // 80 KB

  for (int l = 0; l < 2; ++l) {
    const __bf16* X0 = (l == 0) ? xbf0 : hs0;
    const __bf16* X1 = (l == 0) ? xbf1 : hs1;
    // xg = X * Wih[l]^T + (b_ih+b_hh): grid (N/128=8, M/128=512, 2 inputs)
    gemm_xg<<<dim3(8, 512, 2), 256, 0, stream>>>(
        X0, X1, wih_bf + (size_t)l * G4 * HH, bsum + (size_t)l * G4, xg0, xg1);
    // recurrent scan: grid (8 batch tiles, 2 inputs)
    lstm_scan<<<dim3(8, 2), 256, smem, stream>>>(
        xg0, xg1, whh_bf + (size_t)l * G4 * HH, hs0, hs1,
        hn0 + (size_t)l * BB * HH, hn1 + (size_t)l * BB * HH,
        /*write_seq=*/(l == 0) ? 1 : 0);
  }

  fc_head<<<10, 256, 0, stream>>>(hn0, hn1, fcw, fcb, (float*)d_out);
}
